// ResNet_SNN_expVSL_9672266350611
// MI455X (gfx1250) — compile-verified, matrix operand order checked by reading, not run on hardware
//
#include <hip/hip_runtime.h>
#include <stdint.h>

typedef __attribute__((ext_vector_type(16))) _Float16 v16h;
typedef __attribute__((ext_vector_type(8)))  _Float16 v8h;
typedef __attribute__((ext_vector_type(8)))  float    v8f;
typedef unsigned __attribute__((ext_vector_type(4))) u32x4;
typedef int      __attribute__((ext_vector_type(4))) i32x4;
typedef int      __attribute__((ext_vector_type(8))) i32x8;

#define BATCH   8192
#define TSTEPS  24
#define DECAY_V 0.9f     // 1 - DT*TAU_MEM_INV
#define ALPHA_V 0.1f     // DT*TAU_MEM_INV
#define DECAY_I 0.8f     // 1 - DT*TAU_SYN_INV
#define VTH_CELL 0.085f
#define VTH_ENC  1.0f

// ---------------------------------------------------------------------------
// Kernel 0: convert/pad a weight matrix to f16 [Npad][Kpad] in workspace.
// ---------------------------------------------------------------------------
__global__ __launch_bounds__(256) void wconv_kernel(const float* __restrict__ src,
                                                    _Float16* __restrict__ dst,
                                                    int N, int K, int Kpad, int Npad) {
  const int idx = blockIdx.x * blockDim.x + threadIdx.x;
  if (idx >= Npad * Kpad) return;
  const int n = idx / Kpad, k = idx % Kpad;
  dst[idx] = (_Float16)((n < N && k < K) ? src[(size_t)n * K + k] : 0.0f);
}

// ---------------------------------------------------------------------------
// Kernel 1: spike-latency encoder -> bit-packed spikes enc[t][b][13 words]
// ---------------------------------------------------------------------------
__global__ __launch_bounds__(128) void enc_kernel(const float* __restrict__ x,
                                                  unsigned* __restrict__ enc) {
  const int lane = threadIdx.x & 31;
  const int wid  = (blockIdx.x * blockDim.x + threadIdx.x) >> 5;
  const int b    = wid / 13;
  const int word = wid % 13;
  const int f    = word * 32 + lane;
  const float I  = (f < 400) ? x[(size_t)b * 400 + f] : 0.0f;
  float v = 0.0f;
  bool spiked = false;
  for (int t = 0; t < TSTEPS; ++t) {
    v = DECAY_V * v + ALPHA_V * I;
    const bool z = v > VTH_ENC;
    if (z) v = 0.0f;
    const bool outb = z && !spiked;
    spiked = spiked || z;
    const unsigned m = __builtin_amdgcn_ballot_w32(outb);
    if (lane == 0) enc[((size_t)t * BATCH + b) * 13 + word] = m;
  }
}

// ---------------------------------------------------------------------------
// Layer kernel: A (bit-plane spikes, values 0..NP) x W^T via v_wmma f16->f32.
// Workgroup = 4 waves x 16-row M-tiles (64 batch rows), NTC N-tiles starting
// at N-tile nb0 + blockIdx.y*NTC. Weight slice staged into LDS via the TDM.
// ---------------------------------------------------------------------------
template <int KW, int NTC, int NP, bool IS_LI>
__global__ __launch_bounds__(128) void layer_kernel(
    const _Float16* __restrict__ Wh, int nb0, int wN,
    const unsigned* __restrict__ p0, const unsigned* __restrict__ p1,
    const unsigned* __restrict__ p2,
    unsigned* __restrict__ sout, float* __restrict__ out) {
  constexpr int Kpad = KW * 32;
  constexpr int ROWS = NTC * 16;
  extern __shared__ char smem[];
  _Float16* wt  = (_Float16*)smem;                              // ROWS*Kpad halves
  unsigned* bal = (unsigned*)(smem + (size_t)ROWS * Kpad * 2);  // 4 waves * NTC*8

  const int nbase = nb0 + blockIdx.y * NTC;

  // ---- TDM: DMA the weight slice (2-D tile, 2-byte elements) into LDS ----
  if (threadIdx.x < 32) {
    const unsigned long long ga =
        (unsigned long long)(Wh + (size_t)nbase * 16 * Kpad);
    u32x4 g0;
    g0[0] = 1u;                                  // count=1 (valid descriptor)
    g0[1] = 0u;                                  // lds_addr = 0 (start of dyn LDS)
    g0[2] = (unsigned)ga;                        // global_addr[31:0]
    g0[3] = (unsigned)((ga >> 32) & 0x01FFFFFFu) | 0x80000000u; // addr|type=2
    unsigned g1w[8];
    g1w[0] = 0x00010000u;                        // wg_mask=0, data_size=1 (2B)
    g1w[1] = (unsigned)(Kpad & 0xFFFF) << 16;    // tensor_dim0[15:0]
    g1w[2] = (unsigned)((Kpad >> 16) & 0xFFFF) | ((unsigned)(ROWS & 0xFFFF) << 16);
    g1w[3] = (unsigned)(Kpad & 0xFFFF) << 16;    // tile_dim0 = Kpad
    g1w[4] = (unsigned)(ROWS & 0xFFFF);          // tile_dim1 = ROWS
    g1w[5] = (unsigned)Kpad;                     // tensor_dim0_stride lo32
    g1w[6] = 0u; g1w[7] = 0u;
    i32x8 g1;
#pragma unroll
    for (int i = 0; i < 8; ++i) g1[i] = (int)g1w[i];
    const i32x4 z4 = {0, 0, 0, 0};
    const i32x8 z8 = {0, 0, 0, 0, 0, 0, 0, 0};
    __builtin_amdgcn_tensor_load_to_lds(g0, g1, z4, z4, z8, 0);
    __builtin_amdgcn_s_wait_tensorcnt(0);
  }
  __syncthreads();

  const int wave    = threadIdx.x >> 5;
  const int lane    = threadIdx.x & 31;
  const int m       = lane & 15;
  const int koff    = (lane >> 4) * 8;
  const int rowbase = blockIdx.x * 64 + wave * 16;
  const int brow    = rowbase + m;

  v8f vst[NTC], ist[NTC], vmax[NTC];
#pragma unroll
  for (int i = 0; i < NTC; ++i)
#pragma unroll
    for (int j = 0; j < 8; ++j) {
      vst[i][j] = 0.0f; ist[i][j] = 0.0f; vmax[i][j] = -3.0e38f;
    }

  for (int t = 0; t < TSTEPS; ++t) {
    v8f acc[NTC];
#pragma unroll
    for (int i = 0; i < NTC; ++i)
#pragma unroll
      for (int j = 0; j < 8; ++j) acc[i][j] = 0.0f;

    const size_t roff = ((size_t)t * BATCH + brow) * KW;
    const unsigned* pr0 = p0 + roff;
    const unsigned* pr1 = (NP > 1) ? (p1 + roff) : pr0;
    const unsigned* pr2 = (NP > 2) ? (p2 + roff) : pr0;

#pragma unroll
    for (int kw = 0; kw < KW; ++kw) {
      const unsigned w0 = pr0[kw];
      const unsigned w1 = (NP > 1) ? pr1[kw] : 0u;
      const unsigned w2 = (NP > 2) ? pr2[kw] : 0u;
      const unsigned lo0 = (w0 >> koff) & 0xFFu, hi0 = (w0 >> (koff + 16)) & 0xFFu;
      const unsigned lo1 = (w1 >> koff) & 0xFFu, hi1 = (w1 >> (koff + 16)) & 0xFFu;
      const unsigned lo2 = (w2 >> koff) & 0xFFu, hi2 = (w2 >> (koff + 16)) & 0xFFu;
      v16h a;
#pragma unroll
      for (int i = 0; i < 8; ++i) {
        int clo = (int)((lo0 >> i) & 1u);
        int chi = (int)((hi0 >> i) & 1u);
        if (NP > 1) { clo += (int)((lo1 >> i) & 1u); chi += (int)((hi1 >> i) & 1u); }
        if (NP > 2) { clo += (int)((lo2 >> i) & 1u); chi += (int)((hi2 >> i) & 1u); }
        a[i]     = (_Float16)clo;   // K = kw*32 + koff + i
        a[8 + i] = (_Float16)chi;   // K = kw*32 + koff + 16 + i
      }
#pragma unroll
      for (int nt = 0; nt < NTC; ++nt) {
        const _Float16* wrow = wt + (size_t)(nt * 16 + m) * Kpad + kw * 32 + koff;
        const v8h blo = *(const v8h*)wrow;
        const v8h bhi = *(const v8h*)(wrow + 16);
        const v16h b = __builtin_shufflevector(blo, bhi,
            0, 1, 2, 3, 4, 5, 6, 7, 8, 9, 10, 11, 12, 13, 14, 15);
        acc[nt] = __builtin_amdgcn_wmma_f32_16x16x32_f16(
            false, a, false, b, (short)0, acc[nt], false, false);
      }
    }

    if constexpr (!IS_LI) {
      unsigned* mybal = bal + wave * (NTC * 8);
#pragma unroll
      for (int nt = 0; nt < NTC; ++nt)
#pragma unroll
        for (int j = 0; j < 8; ++j) {
          const float vd = DECAY_V * vst[nt][j] + ALPHA_V * ist[nt][j];
          const float id = DECAY_I * ist[nt][j];
          const bool  s  = vd > VTH_CELL;
          vst[nt][j] = s ? 0.0f : vd;
          ist[nt][j] = id + acc[nt][j];
          const unsigned bm = __builtin_amdgcn_ballot_w32(s);
          if (lane == 0) mybal[nt * 8 + j] = bm;  // [15:0]=row j, [31:16]=row j+8
        }
      asm volatile("s_wait_dscnt 0" ::: "memory");  // wave-local LDS RAW
      constexpr int WORDS = (NTC >= 2) ? NTC / 2 : 1;
#pragma unroll
      for (int e = lane; e < 16 * WORDS; e += 32) {
        const int r = e & 15, w = e >> 4;
        const unsigned b0 = mybal[(2 * w) * 8 + (r & 7)];
        const unsigned b1 = mybal[(2 * w + 1) * 8 + (r & 7)];
        const unsigned word = (r < 8)
            ? ((b0 & 0xFFFFu) | (b1 << 16))
            : ((b0 >> 16) | (b1 & 0xFFFF0000u));
        sout[((size_t)t * BATCH + rowbase + r) * 4 + (nbase >> 1) + w] = word;
      }
    } else {
#pragma unroll
      for (int nt = 0; nt < NTC; ++nt)
#pragma unroll
        for (int j = 0; j < 8; ++j) {
          const float vn = DECAY_V * vst[nt][j] + ALPHA_V * ist[nt][j];
          ist[nt][j] = DECAY_I * ist[nt][j] + acc[nt][j];
          vst[nt][j] = vn;
          vmax[nt][j] = fmaxf(vmax[nt][j], vn);
        }
    }
  }

  if constexpr (IS_LI) {
#pragma unroll
    for (int nt = 0; nt < NTC; ++nt) {
      const int c = (nbase + nt) * 16 + m;
      if (c < wN) {
#pragma unroll
        for (int j = 0; j < 8; ++j) {
          const int r = (lane < 16) ? j : j + 8;   // C/D VGPR row layout
          out[(size_t)(rowbase + r) * wN + c] = vmax[nt][j];
        }
      }
    }
  }
}

// ---------------------------------------------------------------------------
extern "C" void kernel_launch(void* const* d_in, const int* in_sizes, int n_in,
                              void* d_out, int out_size, void* d_ws, size_t ws_size,
                              hipStream_t stream) {
  (void)in_sizes; (void)n_in; (void)out_size; (void)ws_size;
  const float* x     = (const float*)d_in[0];
  const float* w_lin = (const float*)d_in[1];
  const float* w1    = (const float*)d_in[2];
  const float* w2    = (const float*)d_in[3];
  const float* w_li  = (const float*)d_in[4];
  float* out = (float*)d_out;

  unsigned* enc = (unsigned*)d_ws;
  unsigned* s0  = enc + (size_t)TSTEPS * BATCH * 13;
  unsigned* s1  = s0 + (size_t)TSTEPS * BATCH * 4;
  unsigned* s2  = s1 + (size_t)TSTEPS * BATCH * 4;
  _Float16* wlin_h = (_Float16*)(s2 + (size_t)TSTEPS * BATCH * 4);
  _Float16* w1_h   = wlin_h + (size_t)128 * 416;
  _Float16* w2_h   = w1_h   + (size_t)128 * 128;
  _Float16* wli_h  = w2_h   + (size_t)128 * 128;      // [112][128]

  wconv_kernel<<<(128 * 416 + 255) / 256, 256, 0, stream>>>(w_lin, wlin_h, 128, 400, 416, 128);
  wconv_kernel<<<(128 * 128 + 255) / 256, 256, 0, stream>>>(w1, w1_h, 128, 128, 128, 128);
  wconv_kernel<<<(128 * 128 + 255) / 256, 256, 0, stream>>>(w2, w2_h, 128, 128, 128, 128);
  wconv_kernel<<<(112 * 128 + 255) / 256, 256, 0, stream>>>(w_li, wli_h, 101, 128, 128, 112);

  enc_kernel<<<(BATCH * 13) / 4, 128, 0, stream>>>(x, enc);

  const size_t lds0 = (size_t)64 * 416 * 2 + 4 * 32 * 4;   // ~53.8 KB
  const size_t lds1 = (size_t)64 * 128 * 2 + 4 * 32 * 4;   // ~16.9 KB
  const size_t ldsB = (size_t)48 * 128 * 2 + 4 * 24 * 4;

  // lif0: enc (1 plane, K=416) x w_lin          [M-split x N-split grid]
  layer_kernel<13, 4, 1, false><<<dim3(BATCH / 64, 2), 128, lds0, stream>>>(
      wlin_h, 0, 128, enc, nullptr, nullptr, s0, nullptr);
  // lif1: s0 x w1
  layer_kernel<4, 4, 1, false><<<dim3(BATCH / 64, 2), 128, lds1, stream>>>(
      w1_h, 0, 128, s0, nullptr, nullptr, s1, nullptr);
  // lif2: (s0+s1) x w2  (residual as summed bit-planes)
  layer_kernel<4, 4, 2, false><<<dim3(BATCH / 64, 2), 128, lds1, stream>>>(
      w2_h, 0, 128, s0, s1, nullptr, s2, nullptr);
  // li: (s0+s1+s2) x w_li, running max over t -> out [8192,101]
  layer_kernel<4, 4, 3, true><<<dim3(BATCH / 64, 1), 128, lds1, stream>>>(
      wli_h, 0, 101, s0, s1, s2, nullptr, out);
  layer_kernel<4, 3, 3, true><<<dim3(BATCH / 64, 1), 128, ldsB, stream>>>(
      wli_h, 4, 101, s0, s1, s2, nullptr, out);
}
